// JointMultipleKernelMaximumMeanDiscrepancy_81698867904838
// MI455X (gfx1250) — matrix operand order repaired and checked by reading, not compiled, not run on hardware
//
#include <hip/hip_runtime.h>
#include <math.h>

// Problem constants (from reference setup_inputs)
#define BHALF 2048            // b
#define NTOT  4096            // 2b
#define D1    4096
#define D2    2048

#define TILE    128
#define KSTEP   32
#define LDS_STR 40            // shorts per LDS tile row (32 + 8 pad), 80B => 16B aligned
#define NTILES  (NTOT / TILE) // 32
#define NBLOCKS (NTILES * (NTILES + 1) / 2)  // 528 lower-tri tiles

typedef __attribute__((ext_vector_type(16))) __bf16 v16bf;
typedef __attribute__((ext_vector_type(8)))  float  v8f;
typedef int v4i_vs __attribute__((vector_size(16)));   // matches async-LDS builtin param

// CDNA5 async global->LDS DMA path (ASYNCcnt), if the toolchain exposes it
#if defined(__has_builtin)
#  if __has_builtin(__builtin_amdgcn_global_load_async_to_lds_b128) && \
      __has_builtin(__builtin_amdgcn_s_wait_asynccnt)
#    define HAVE_ASYNC_LDS 1
#  endif
#endif
#ifndef HAVE_ASYNC_LDS
#  define HAVE_ASYNC_LDS 0
#endif

#if HAVE_ASYNC_LDS
#  define NBUF 2
#else
#  define NBUF 1
#endif

// workspace layout: float scratch region, then bf16 hi/lo planes
#define WS_SQ1  0
#define WS_SQ2  (WS_SQ1 + NTOT)
#define WS_CS1  (WS_SQ2 + NTOT)
#define WS_CS2  (WS_CS1 + D1)
#define WS_BW   (WS_CS2 + D2)
#define WS_PART (WS_BW + 8)

#define N1ELEM  ((size_t)NTOT * (size_t)D1)   // 16.78M
#define N2ELEM  ((size_t)NTOT * (size_t)D2)   //  8.39M
#define WS_PLANES_OFF ((size_t)65536)          // bytes; float region ends < 60KB
#define WS_NEEDED (WS_PLANES_OFF + 2 * 2 * N1ELEM + 2 * 2 * N2ELEM)  // ~96.1 MB

__device__ __forceinline__ const float* row_ptr(const float* s, const float* t, int row, int d) {
  return (row < BHALF) ? (s + (size_t)row * d) : (t + (size_t)(row - BHALF) * d);
}

__device__ __forceinline__ unsigned short f32_to_bf16(float f) {
  unsigned u = __float_as_uint(f);
  unsigned r = 0x7FFFu + ((u >> 16) & 1u);   // round-to-nearest-even
  return (unsigned short)((u + r) >> 16);
}
__device__ __forceinline__ float bf16_to_f32(unsigned short h) {
  return __uint_as_float(((unsigned)h) << 16);
}

// 5-kernel Gaussian sum: sum_i exp(-l2 / (bw * 2^i)), invbw = -1/bw
__device__ __forceinline__ float gauss5(float l2, float invbw) {
  float e = l2 * invbw;
  return __expf(e) + __expf(e * 0.5f) + __expf(e * 0.25f)
       + __expf(e * 0.125f) + __expf(e * 0.0625f);
}

struct Frag { union { uint4 u[2]; v16bf v; }; };
struct TileBuf { unsigned short *ahi, *alo, *bhi, *blo; };

// ---------------------------------------------------------------------------
// Pass A0: one-shot f32 -> (bf16 hi, bf16 lo) plane conversion
// ---------------------------------------------------------------------------
__global__ void __launch_bounds__(256)
preconvert_kernel(const float* z1s, const float* z1t, const float* z2s, const float* z2t,
                  unsigned short* t1hi, unsigned short* t1lo,
                  unsigned short* t2hi, unsigned short* t2lo) {
  size_t e = ((size_t)blockIdx.x * 256 + threadIdx.x) * 8;   // 8 elems/thread
  const float* rp; unsigned short* hi; unsigned short* lo;
  if (e < N1ELEM) {
    size_t row = e / D1, col = e % D1;
    rp = row_ptr(z1s, z1t, (int)row, D1) + col;
    hi = t1hi + e; lo = t1lo + e;
  } else {
    size_t e2 = e - N1ELEM;
    size_t row = e2 / D2, col = e2 % D2;
    rp = row_ptr(z2s, z2t, (int)row, D2) + col;
    hi = t2hi + e2; lo = t2lo + e2;
  }
  float4 v0 = *(const float4*)(rp);
  float4 v1 = *(const float4*)(rp + 4);
  float x[8] = { v0.x, v0.y, v0.z, v0.w, v1.x, v1.y, v1.z, v1.w };
  union { unsigned short s[8]; uint4 u; } H, L;
#pragma unroll
  for (int k = 0; k < 8; ++k) {
    unsigned short h = f32_to_bf16(x[k]);
    H.s[k] = h;
    L.s[k] = f32_to_bf16(x[k] - bf16_to_f32(h));
  }
  *(uint4*)hi = H.u;
  *(uint4*)lo = L.u;
}

// ---------------------------------------------------------------------------
// Pass A1: per-row squared norms (f32 inputs, exact)
// ---------------------------------------------------------------------------
__global__ void __launch_bounds__(256)
rownorm_kernel(const float* z1s, const float* z1t, const float* z2s, const float* z2t,
               float* sq1, float* sq2) {
  __shared__ float red[256];
  int rb = blockIdx.x, tid = threadIdx.x;
  const float* rp; int d; float* out;
  if (rb < NTOT) { d = D1; rp = row_ptr(z1s, z1t, rb, D1); out = sq1 + rb; }
  else { int r = rb - NTOT; d = D2; rp = row_ptr(z2s, z2t, r, D2); out = sq2 + r; }
  float s = 0.f;
  for (int c = tid; c < d; c += 256) { float v = rp[c]; s += v * v; }
  red[tid] = s; __syncthreads();
  for (int k = 128; k > 0; k >>= 1) { if (tid < k) red[tid] += red[tid + k]; __syncthreads(); }
  if (tid == 0) *out = red[0];
}

// ---------------------------------------------------------------------------
// Pass A2: column sums (sum(l2) = 2n*sumsq - 2*||colsum||^2, no GEMM needed)
// ---------------------------------------------------------------------------
__global__ void __launch_bounds__(256)
colsum_kernel(const float* z1s, const float* z1t, const float* z2s, const float* z2t,
              float* cs1, float* cs2) {
  int tid = threadIdx.x;
  if (blockIdx.x < D1 / 256) {
    int c = blockIdx.x * 256 + tid;
    float s = 0.f;
    for (int r = 0; r < NTOT; ++r) s += row_ptr(z1s, z1t, r, D1)[c];
    cs1[c] = s;
  } else {
    int c = (blockIdx.x - D1 / 256) * 256 + tid;
    float s = 0.f;
    for (int r = 0; r < NTOT; ++r) s += row_ptr(z2s, z2t, r, D2)[c];
    cs2[c] = s;
  }
}

// ---------------------------------------------------------------------------
// Pass A3: bandwidths (single block, fixed-order reductions => deterministic)
// ---------------------------------------------------------------------------
__global__ void __launch_bounds__(256)
bw_kernel(const float* sq1, const float* sq2, const float* cs1, const float* cs2, float* bw) {
  __shared__ float red[256];
  int tid = threadIdx.x;
  float a = 0.f, b = 0.f, c = 0.f, d = 0.f;
  for (int i = tid; i < NTOT; i += 256) { a += sq1[i]; b += sq2[i]; }
  for (int i = tid; i < D1; i += 256) { float v = cs1[i]; c += v * v; }
  for (int i = tid; i < D2; i += 256) { float v = cs2[i]; d += v * v; }
  float vals[4] = { a, b, c, d }; float out[4];
  for (int k = 0; k < 4; ++k) {
    red[tid] = vals[k]; __syncthreads();
    for (int s = 128; s > 0; s >>= 1) { if (tid < s) red[tid] += red[tid + s]; __syncthreads(); }
    out[k] = red[0]; __syncthreads();
  }
  if (tid == 0) {
    const double n = (double)NTOT;
    double sumL2_1 = 2.0 * n * (double)out[0] - 2.0 * (double)out[2];
    double sumL2_2 = 2.0 * n * (double)out[1] - 2.0 * (double)out[3];
    double denom = n * n - n;
    bw[0] = (float)(sumL2_1 / denom / 4.0);   // /= KERNEL_MUL^(KERNEL_NUM//2)
    bw[1] = (float)(sumL2_2 / denom / 4.0);
  }
}

// ---------------------------------------------------------------------------
// Shared WMMA inner step: 2x4 fragments/wave, bf16 hi/lo split (3 WMMAs)
// ---------------------------------------------------------------------------
__device__ __forceinline__ void mma_step(const unsigned short* ahi, const unsigned short* alo,
                                         const unsigned short* bhi, const unsigned short* blo,
                                         int wm, int wn, int lane, v8f (&acc)[2][4]) {
  int half = lane >> 4, lrow = lane & 15;
  // A fragment (16-bit, 16x32): lanes 0-15: M=lane, K in {0..7,16..23};
  // lanes 16-31: M=lane-16, K in {8..15,24..31}.
  Frag aH[2], aL[2];
#pragma unroll
  for (int mf = 0; mf < 2; ++mf) {
    int row = wm * 32 + mf * 16 + lrow;
    const unsigned short* p = ahi + row * LDS_STR;
    const unsigned short* q = alo + row * LDS_STR;
    int o0 = half * 8, o1 = 16 + half * 8;
    aH[mf].u[0] = *(const uint4*)(p + o0); aH[mf].u[1] = *(const uint4*)(p + o1);
    aL[mf].u[0] = *(const uint4*)(q + o0); aL[mf].u[1] = *(const uint4*)(q + o1);
  }
  // B fragment (16-bit, 32x16): lanes 0-15: N=lane, K in {0..7,16..23}; lanes 16-31: N=lane-16, K in {8..15,24..31}.
  // B = T^T tile => B[k][n] = Trow(n)[k] = 32 contiguous bf16 of LDS row n.
#pragma unroll
  for (int nf = 0; nf < 4; ++nf) {
    int row = wn * 64 + nf * 16 + lrow;
    const unsigned short* p = bhi + row * LDS_STR + half * 8;
    const unsigned short* q = blo + row * LDS_STR + half * 8;
    Frag bH, bL;
    bH.u[0] = *(const uint4*)(p);     bH.u[1] = *(const uint4*)(p + 16);
    bL.u[0] = *(const uint4*)(q);     bL.u[1] = *(const uint4*)(q + 16);
#pragma unroll
    for (int mf = 0; mf < 2; ++mf) {
      v8f c = acc[mf][nf];
      // x*y ~= xh*yh + xh*yl + xl*yh  (drop xl*yl), f32 accumulate
      c = __builtin_amdgcn_wmma_f32_16x16x32_bf16(false, aH[mf].v, false, bH.v, (short)0, c, false, false);
      c = __builtin_amdgcn_wmma_f32_16x16x32_bf16(false, aH[mf].v, false, bL.v, (short)0, c, false, false);
      c = __builtin_amdgcn_wmma_f32_16x16x32_bf16(false, aL[mf].v, false, bH.v, (short)0, c, false, false);
      acc[mf][nf] = c;
    }
  }
}

#if HAVE_ASYNC_LDS
// ---------------------------------------------------------------------------
// Fast path staging via CDNA5 async global->LDS DMA (no VGPR data, ASYNCcnt)
// builtin signature: (v4i AS1* gaddr, v4i AS3* ldsaddr, imm offset, imm cpol)
// ---------------------------------------------------------------------------
__device__ __forceinline__ void async_cp16(const unsigned short* g, unsigned short* l) {
  __builtin_amdgcn_global_load_async_to_lds_b128(
      (v4i_vs __attribute__((address_space(1)))*)(g),
      (v4i_vs __attribute__((address_space(3)))*)(l), 0, 0);
}

// 8 async b128 issues per thread (2 chunks x 4 planes)
__device__ __forceinline__ void async_tile_copy(
    const unsigned short* __restrict__ hiP, const unsigned short* __restrict__ loP,
    int iBase, int jBase, int d, int k0, TileBuf b) {
  int tid = threadIdx.x;
#pragma unroll
  for (int q = 0; q < 2; ++q) {
    int idx = tid + q * 256;            // 0..511 = 128 rows x 4 uint4
    int r = idx >> 2, c = idx & 3;
    size_t offA = (size_t)(iBase + r) * d + k0 + c * 8;
    size_t offB = (size_t)(jBase + r) * d + k0 + c * 8;
    int so = r * LDS_STR + c * 8;
    async_cp16(hiP + offA, b.ahi + so);
    async_cp16(loP + offA, b.alo + so);
    async_cp16(hiP + offB, b.bhi + so);
    async_cp16(loP + offB, b.blo + so);
  }
}

__device__ __forceinline__ void gram_pass(
    const unsigned short* __restrict__ hiP, const unsigned short* __restrict__ loP,
    int d, int iBase, int jBase, int wm, int wn, int lane,
    TileBuf b0, TileBuf b1, v8f (&acc)[2][4]) {
  async_tile_copy(hiP, loP, iBase, jBase, d, 0, b0);
  int nsteps = d / KSTEP;
  for (int s = 0; s < nsteps; ++s) {
    TileBuf cur = (s & 1) ? b1 : b0;
    TileBuf nxt = (s & 1) ? b0 : b1;
    if (s + 1 < nsteps) {
      async_tile_copy(hiP, loP, iBase, jBase, d, (s + 1) * KSTEP, nxt);
      __builtin_amdgcn_s_wait_asynccnt(8);   // async loads retire in order: cur landed
    } else {
      __builtin_amdgcn_s_wait_asynccnt(0);
    }
    __syncthreads();                          // all waves' DMA visible
    mma_step(cur.ahi, cur.alo, cur.bhi, cur.blo, wm, wn, lane, acc);
    __syncthreads();                          // reads done before buffer reuse
  }
}
#else
// ---------------------------------------------------------------------------
// Fast path staging via registers (fallback when async builtins are absent)
// ---------------------------------------------------------------------------
struct Stage { uint4 v[2][2][2]; };  // [tile A/B][plane hi/lo][q]

__device__ __forceinline__ void stage_load(Stage& st,
    const unsigned short* __restrict__ hiP, const unsigned short* __restrict__ loP,
    int iBase, int jBase, int d, int k0) {
  int tid = threadIdx.x;
#pragma unroll
  for (int q = 0; q < 2; ++q) {
    int idx = tid + q * 256;
    int r = idx >> 2, c = idx & 3;
    size_t offA = (size_t)(iBase + r) * d + k0 + c * 8;
    size_t offB = (size_t)(jBase + r) * d + k0 + c * 8;
    st.v[0][0][q] = *(const uint4*)(hiP + offA);
    st.v[0][1][q] = *(const uint4*)(loP + offA);
    st.v[1][0][q] = *(const uint4*)(hiP + offB);
    st.v[1][1][q] = *(const uint4*)(loP + offB);
  }
}

__device__ __forceinline__ void stage_store(const Stage& st, TileBuf b) {
  int tid = threadIdx.x;
#pragma unroll
  for (int q = 0; q < 2; ++q) {
    int idx = tid + q * 256;
    int r = idx >> 2, c = idx & 3;
    int so = r * LDS_STR + c * 8;
    *(uint4*)(b.ahi + so) = st.v[0][0][q];
    *(uint4*)(b.alo + so) = st.v[0][1][q];
    *(uint4*)(b.bhi + so) = st.v[1][0][q];
    *(uint4*)(b.blo + so) = st.v[1][1][q];
  }
}

__device__ __forceinline__ void gram_pass(
    const unsigned short* __restrict__ hiP, const unsigned short* __restrict__ loP,
    int d, int iBase, int jBase, int wm, int wn, int lane,
    TileBuf b0, TileBuf b1 /*unused*/, v8f (&acc)[2][4]) {
  Stage st;
  stage_load(st, hiP, loP, iBase, jBase, d, 0);
  for (int k0 = 0; k0 < d; k0 += KSTEP) {
    stage_store(st, b0);
    __syncthreads();
    if (k0 + KSTEP < d) stage_load(st, hiP, loP, iBase, jBase, d, k0 + KSTEP);
    mma_step(b0.ahi, b0.alo, b0.bhi, b0.blo, wm, wn, lane, acc);
    __syncthreads();
  }
}
#endif // HAVE_ASYNC_LDS

// Phase-1 epilogue: turn Gram-1 accumulators into k1 values, stash in LDS.
// Each lane later reads back exactly the elements it wrote -> no barrier needed.
__device__ __forceinline__ void kernel1_to_lds(v8f (&acc)[2][4], int iBase, int jBase,
    int wm, int wn, int lane, const float* __restrict__ ws_sq, float invbw, float* k1buf) {
  int half = lane >> 4, lrow = lane & 15;
#pragma unroll
  for (int mf = 0; mf < 2; ++mf) {
#pragma unroll
    for (int nf = 0; nf < 4; ++nf) {
      int nRel = wn * 64 + nf * 16 + lrow;
      float sqj = ws_sq[jBase + nRel];
#pragma unroll
      for (int r = 0; r < 8; ++r) {
        int mRel = wm * 32 + mf * 16 + half * 8 + r;  // C layout: VGPR r -> M=r / r+8
        float l2 = fmaxf(ws_sq[iBase + mRel] + sqj - 2.0f * acc[mf][nf][r], 0.f);
        k1buf[mRel * TILE + nRel] = gauss5(l2, invbw);
      }
    }
  }
}

// Phase-2 epilogue: k2 from Gram-2 acc, k1 from LDS, index weight, block partial.
__device__ __forceinline__ void final_accum(v8f (&acc)[2][4], int iBase, int jBase,
    int ti, int tj, int wm, int wn, int lane,
    const float* __restrict__ ws_sq, float invbw, const float* k1buf,
    float* sred, float* __restrict__ partials, int p) {
  const float offw = 1.0f / ((float)BHALF * (float)(BHALF - 1));
  const float negw = -1.0f / ((float)BHALF * (float)BHALF);
  int half = lane >> 4, lrow = lane & 15;
  int tid = threadIdx.x;
  float local = 0.f;
#pragma unroll
  for (int mf = 0; mf < 2; ++mf) {
#pragma unroll
    for (int nf = 0; nf < 4; ++nf) {
      int nRel = wn * 64 + nf * 16 + lrow;
      int j = jBase + nRel;
      float sqj = ws_sq[j];
#pragma unroll
      for (int r = 0; r < 8; ++r) {
        int mRel = wm * 32 + mf * 16 + half * 8 + r;
        int i = iBase + mRel;
        float l2 = fmaxf(ws_sq[i] + sqj - 2.0f * acc[mf][nf][r], 0.f);
        float k2 = gauss5(l2, invbw);
        float k1 = k1buf[mRel * TILE + nRel];
        bool is = i < BHALF, js = j < BHALF;
        float w = (is == js) ? ((i == j) ? 0.f : offw) : negw;
        local += w * k1 * k2;
      }
    }
  }
  local *= (ti == tj) ? 1.f : 2.f;   // lower-tri symmetry
  sred[tid] = local; __syncthreads();
  for (int s = 128; s > 0; s >>= 1) { if (tid < s) sred[tid] += sred[tid + s]; __syncthreads(); }
  if (tid == 0) partials[p] = sred[0];
}

__device__ __forceinline__ void tri_index(int p, int& ti, int& tj) {
  ti = (int)((sqrtf(8.f * (float)p + 1.f) - 1.f) * 0.5f);
  while ((ti + 1) * (ti + 2) / 2 <= p) ++ti;
  while (ti * (ti + 1) / 2 > p) --ti;
  tj = p - ti * (ti + 1) / 2;
}

__global__ void
__attribute__((amdgpu_flat_work_group_size(256, 256), amdgpu_waves_per_eu(1)))
jmmd_tile_pre(const unsigned short* __restrict__ t1hi, const unsigned short* __restrict__ t1lo,
              const unsigned short* __restrict__ t2hi, const unsigned short* __restrict__ t2lo,
              const float* __restrict__ ws_sq1, const float* __restrict__ ws_sq2,
              const float* __restrict__ ws_bw, float* __restrict__ partials) {
  __shared__ unsigned short sT[NBUF][4][TILE * LDS_STR];  // staging planes
  __shared__ float k1buf[TILE * TILE];                    // 64 KB: k1 between passes
  __shared__ float sred[256];

  TileBuf b0 = { sT[0][0], sT[0][1], sT[0][2], sT[0][3] };
  TileBuf b1 = { sT[NBUF - 1][0], sT[NBUF - 1][1], sT[NBUF - 1][2], sT[NBUF - 1][3] };

  int p = blockIdx.x, ti, tj;
  tri_index(p, ti, tj);
  int iBase = ti * TILE, jBase = tj * TILE;
  int tid = threadIdx.x, wave = tid >> 5, lane = tid & 31;
  int wm = wave & 3, wn = wave >> 2;
  float inv1 = -1.0f / ws_bw[0], inv2 = -1.0f / ws_bw[1];
  v8f zero = { 0.f, 0.f, 0.f, 0.f, 0.f, 0.f, 0.f, 0.f };

  {                                     // pass 1: acc scoped -> dies before pass 2
    v8f acc[2][4];
#pragma unroll
    for (int a = 0; a < 2; ++a)
#pragma unroll
      for (int c = 0; c < 4; ++c) acc[a][c] = zero;
    gram_pass(t1hi, t1lo, D1, iBase, jBase, wm, wn, lane, b0, b1, acc);
    kernel1_to_lds(acc, iBase, jBase, wm, wn, lane, ws_sq1, inv1, k1buf);
  }
  {                                     // pass 2
    v8f acc[2][4];
#pragma unroll
    for (int a = 0; a < 2; ++a)
#pragma unroll
      for (int c = 0; c < 4; ++c) acc[a][c] = zero;
    gram_pass(t2hi, t2lo, D2, iBase, jBase, wm, wn, lane, b0, b1, acc);
    final_accum(acc, iBase, jBase, ti, tj, wm, wn, lane, ws_sq2, inv2, k1buf, sred, partials, p);
  }
}

// ---------------------------------------------------------------------------
// Fallback path (workspace too small): convert f32->bf16 hi/lo on the fly
// ---------------------------------------------------------------------------
__device__ __forceinline__ void load_tile_conv(const float* s, const float* t, int rowBase, int d,
                                               int k0, unsigned short* hi, unsigned short* lo) {
  int tid = threadIdx.x;
#pragma unroll
  for (int q = 0; q < 4; ++q) {
    int idx = tid + q * 256;
    int r = idx >> 3, c4 = idx & 7;
    const float* rp = row_ptr(s, t, rowBase + r, d);
    float4 v = *(const float4*)(rp + k0 + 4 * c4);
    int so = r * LDS_STR + 4 * c4;
    float x[4] = { v.x, v.y, v.z, v.w };
#pragma unroll
    for (int e = 0; e < 4; ++e) {
      unsigned short h = f32_to_bf16(x[e]);
      hi[so + e] = h;
      lo[so + e] = f32_to_bf16(x[e] - bf16_to_f32(h));
    }
  }
}

__global__ void
__attribute__((amdgpu_flat_work_group_size(256, 256), amdgpu_waves_per_eu(1)))
jmmd_tile_conv(const float* z1s, const float* z1t, const float* z2s, const float* z2t,
               const float* __restrict__ ws_sq1, const float* __restrict__ ws_sq2,
               const float* __restrict__ ws_bw, float* __restrict__ partials) {
  __shared__ unsigned short sAhi[TILE * LDS_STR], sAlo[TILE * LDS_STR];
  __shared__ unsigned short sBhi[TILE * LDS_STR], sBlo[TILE * LDS_STR];
  __shared__ float k1buf[TILE * TILE];
  __shared__ float sred[256];

  int p = blockIdx.x, ti, tj;
  tri_index(p, ti, tj);
  int iBase = ti * TILE, jBase = tj * TILE;
  int tid = threadIdx.x, wave = tid >> 5, lane = tid & 31;
  int wm = wave & 3, wn = wave >> 2;
  float inv1 = -1.0f / ws_bw[0], inv2 = -1.0f / ws_bw[1];
  v8f zero = { 0.f, 0.f, 0.f, 0.f, 0.f, 0.f, 0.f, 0.f };

  {
    v8f acc[2][4];
#pragma unroll
    for (int a = 0; a < 2; ++a)
#pragma unroll
      for (int c = 0; c < 4; ++c) acc[a][c] = zero;
    for (int k0 = 0; k0 < D1; k0 += KSTEP) {
      load_tile_conv(z1s, z1t, iBase, D1, k0, sAhi, sAlo);
      load_tile_conv(z1s, z1t, jBase, D1, k0, sBhi, sBlo);
      __syncthreads();
      mma_step(sAhi, sAlo, sBhi, sBlo, wm, wn, lane, acc);
      __syncthreads();
    }
    kernel1_to_lds(acc, iBase, jBase, wm, wn, lane, ws_sq1, inv1, k1buf);
  }
  {
    v8f acc[2][4];
#pragma unroll
    for (int a = 0; a < 2; ++a)
#pragma unroll
      for (int c = 0; c < 4; ++c) acc[a][c] = zero;
    for (int k0 = 0; k0 < D2; k0 += KSTEP) {
      load_tile_conv(z2s, z2t, iBase, D2, k0, sAhi, sAlo);
      load_tile_conv(z2s, z2t, jBase, D2, k0, sBhi, sBlo);
      __syncthreads();
      mma_step(sAhi, sAlo, sBhi, sBlo, wm, wn, lane, acc);
      __syncthreads();
    }
    final_accum(acc, iBase, jBase, ti, tj, wm, wn, lane, ws_sq2, inv2, k1buf, sred, partials, p);
  }
}

// ---------------------------------------------------------------------------
// Final deterministic reduction of per-tile partials
// ---------------------------------------------------------------------------
__global__ void __launch_bounds__(256)
final_kernel(const float* partials, float* out) {
  __shared__ float red[256];
  int tid = threadIdx.x;
  float s = 0.f;
  for (int i = tid; i < NBLOCKS; i += 256) s += partials[i];
  red[tid] = s; __syncthreads();
  for (int k = 128; k > 0; k >>= 1) { if (tid < k) red[tid] += red[tid + k]; __syncthreads(); }
  if (tid == 0) out[0] = red[0] + 2.0f / (float)(BHALF - 1);
}

extern "C" void kernel_launch(void* const* d_in, const int* in_sizes, int n_in,
                              void* d_out, int out_size, void* d_ws, size_t ws_size,
                              hipStream_t stream) {
  const float* z1s = (const float*)d_in[0];
  const float* z1t = (const float*)d_in[1];
  const float* z2s = (const float*)d_in[2];
  const float* z2t = (const float*)d_in[3];
  float* W = (float*)d_ws;
  float* sq1 = W + WS_SQ1;
  float* sq2 = W + WS_SQ2;
  float* cs1 = W + WS_CS1;
  float* cs2 = W + WS_CS2;
  float* bw  = W + WS_BW;
  float* prt = W + WS_PART;

  rownorm_kernel<<<2 * NTOT, 256, 0, stream>>>(z1s, z1t, z2s, z2t, sq1, sq2);
  colsum_kernel<<<(D1 + D2) / 256, 256, 0, stream>>>(z1s, z1t, z2s, z2t, cs1, cs2);
  bw_kernel<<<1, 256, 0, stream>>>(sq1, sq2, cs1, cs2, bw);

  if (ws_size >= WS_NEEDED) {
    // fast path: preconvert once, then DMA-staged WMMA GEMM
    unsigned short* t1hi = (unsigned short*)((char*)d_ws + WS_PLANES_OFF);
    unsigned short* t1lo = t1hi + N1ELEM;
    unsigned short* t2hi = t1lo + N1ELEM;
    unsigned short* t2lo = t2hi + N2ELEM;
    size_t nconv_blocks = (N1ELEM + N2ELEM) / 8 / 256;   // 12288
    preconvert_kernel<<<(int)nconv_blocks, 256, 0, stream>>>(z1s, z1t, z2s, z2t,
                                                             t1hi, t1lo, t2hi, t2lo);
    jmmd_tile_pre<<<NBLOCKS, 256, 0, stream>>>(t1hi, t1lo, t2hi, t2lo, sq1, sq2, bw, prt);
  } else {
    jmmd_tile_conv<<<NBLOCKS, 256, 0, stream>>>(z1s, z1t, z2s, z2t, sq1, sq2, bw, prt);
  }
  final_kernel<<<1, 256, 0, stream>>>(prt, (float*)d_out);
}
